// CausalSelfAttention_87016037417662
// MI455X (gfx1250) — matrix operand order repaired and checked by reading, not compiled, not run on hardware
//
#include <hip/hip_runtime.h>
#include <hip/hip_bf16.h>

typedef unsigned short u16;
typedef unsigned int   u32;

typedef __attribute__((ext_vector_type(8)))  __bf16 v8bf;
typedef __attribute__((ext_vector_type(16))) __bf16 v16bf;
typedef __attribute__((ext_vector_type(8)))  float  v8f;

#define B_DIM   2
#define T_DIM   2048
#define C_DIM   2048
#define NH      16
#define NKV     8
#define HD      128
#define M_ROWS  (B_DIM * T_DIM)     // 4096
#define QKV_N   4096                // 2048 q + 1024 k + 1024 v
#define SCALE_ATTN 0.08838834764831845f  // 1/sqrt(128)

// ---------------- bf16 helpers (bit-exact, no reliance on __bf16 arithmetic) --
__device__ __forceinline__ u16 f2bf(float f) {
  u32 u = __builtin_bit_cast(u32, f);
  u32 r = u + 0x7FFFu + ((u >> 16) & 1u);   // round-to-nearest-even
  return (u16)(r >> 16);
}
__device__ __forceinline__ float bf2f(u16 h) {
  return __builtin_bit_cast(float, (u32)h << 16);
}
__device__ __forceinline__ v8f zero8() {
  v8f z;
#pragma unroll
  for (int i = 0; i < 8; ++i) z[i] = 0.0f;
  return z;
}
__device__ __forceinline__ v8f wmma_bf16(v16bf a, v16bf b, v8f c) {
  return __builtin_amdgcn_wmma_f32_16x16x32_bf16(false, a, false, b, (short)0, c, false, false);
}
// A fragment 16x32 bf16: lane holds row m=lane&15; elems 0..7 -> k = lh*8+e,
// elems 8..15 -> k = 16+lh*8+(e-8).  `row` points at k=0 of this lane's row.
__device__ __forceinline__ v16bf frag_a(const u16* row, int lh) {
  v8bf lo = *reinterpret_cast<const v8bf*>(row + lh * 8);
  v8bf hi = *reinterpret_cast<const v8bf*>(row + 16 + lh * 8);
  return __builtin_shufflevector(lo, hi, 0,1,2,3,4,5,6,7,8,9,10,11,12,13,14,15);
}
// B fragment 32x16 bf16: lane holds col n=lane&15; elems 0..7 -> k = lh*8+e,
// elems 8..15 -> k = 16+lh*8+(e-8) (same order as frag_a).
__device__ __forceinline__ v16bf frag_b(const u16* col, int lh) {
  v8bf lo = *reinterpret_cast<const v8bf*>(col + lh * 8);
  v8bf hi = *reinterpret_cast<const v8bf*>(col + 16 + lh * 8);
  return __builtin_shufflevector(lo, hi, 0,1,2,3,4,5,6,7,8,9,10,11,12,13,14,15);
}
__device__ __forceinline__ float wave_sum(float v) {
#pragma unroll
  for (int m = 16; m; m >>= 1) v += __shfl_xor(v, m, 32);
  return v;
}

// ---------------- CDNA5 async global->LDS copy (ASYNCcnt path) ---------------
// Per-lane DMA: copies 16 bytes from global vaddr to LDS at vdst_addr, no VGPR data movement.
// Low 32 bits of a generic shared pointer are the LDS byte offset (ISA 10.2).
__device__ __forceinline__ void async_copy_b128(u32 lds_byte, const void* gaddr) {
  asm volatile("global_load_async_to_lds_b128 %0, %1, off"
               :: "v"(lds_byte), "v"(gaddr) : "memory");
}
__device__ __forceinline__ void async_wait0() {
  asm volatile("s_wait_asynccnt 0x0" ::: "memory");
}
__device__ __forceinline__ u32 lds_off(const void* p) {
  return (u32)(uintptr_t)p;
}

// ---------------- f32 -> bf16 convert with column placement ------------------
__global__ void cvt_f32_bf16(const float* __restrict__ src, u16* __restrict__ dst,
                             int rows, int cols, int dld, int coff) {
  long i = (long)blockIdx.x * blockDim.x + threadIdx.x;
  if (i >= (long)rows * cols) return;
  int r = (int)(i / cols), c = (int)(i % cols);
  dst[(size_t)r * dld + coff + c] = f2bf(src[i]);
}

// ---------------- WMMA GEMM: C[MxN] = A[MxK](bf16) * B[KxN](bf16), f32 out ----
#define LDA_P 40   // 32 + 8 pad (80B rows, 16B aligned)
#define LDB_P 40
__global__ __launch_bounds__(128) void gemm_bf16(
    const u16* __restrict__ A, const u16* __restrict__ B, float* __restrict__ C,
    int M, int N, int K, const float* __restrict__ scale_ptr) {
  __shared__ __align__(16) u16 sA[128 * LDA_P];
  __shared__ __align__(16) u16 sBt[128 * LDB_P];

  const int tid = threadIdx.x;
  const int wave = tid >> 5, lane = tid & 31;
  const int lh = lane >> 4, nl = lane & 15;
  const int bM = blockIdx.y * 128, bN = blockIdx.x * 128;
  const int wm = (wave >> 1) * 64, wn = (wave & 1) * 64;

  v8f acc[4][4];
#pragma unroll
  for (int i = 0; i < 4; ++i)
#pragma unroll
    for (int j = 0; j < 4; ++j) acc[i][j] = zero8();

  for (int kk = 0; kk < K; kk += 32) {
    {  // A tile: thread t stages its row's 32 elems via async DMA (4x16B)
      const u16* src = A + (size_t)(bM + tid) * K + kk;
      u32 dst = lds_off(&sA[tid * LDA_P]);
#pragma unroll
      for (int i = 0; i < 4; ++i)
        async_copy_b128(dst + i * 16, src + i * 8);
    }
    {  // B tile, stored transposed: sBt[n][k] (transpose -> VGPR path)
      int kr = tid >> 2, cg = (tid & 3) * 32;
      const u16* src = B + (size_t)(kk + kr) * N + bN + cg;
#pragma unroll
      for (int i = 0; i < 32; ++i) sBt[(cg + i) * LDB_P + kr] = src[i];
    }
    async_wait0();
    __syncthreads();

    v16bf bF[4];
#pragma unroll
    for (int j = 0; j < 4; ++j) bF[j] = frag_b(&sBt[(wn + j * 16 + nl) * LDB_P], lh);
#pragma unroll
    for (int i = 0; i < 4; ++i) {
      v16bf aF = frag_a(&sA[(wm + i * 16 + nl) * LDA_P], lh);
#pragma unroll
      for (int j = 0; j < 4; ++j) acc[i][j] = wmma_bf16(aF, bF[j], acc[i][j]);
    }
    __syncthreads();
  }

  float sc = scale_ptr ? (1.0f + scale_ptr[0]) : 1.0f;
#pragma unroll
  for (int i = 0; i < 4; ++i)
#pragma unroll
    for (int j = 0; j < 4; ++j)
#pragma unroll
      for (int r = 0; r < 8; ++r) {
        int gr = bM + wm + i * 16 + r + 8 * lh;
        int gc = bN + wn + j * 16 + nl;
        C[(size_t)gr * N + gc] = acc[i][j][r] * sc;
      }
}

// ---------------- gates + RoPE + k-shift + RMS-norm, write bf16 Q/K/V --------
__global__ __launch_bounds__(256) void qkv_prep(
    const float* __restrict__ x, const float* __restrict__ ve,
    const float* __restrict__ cosp, const float* __restrict__ sinp,
    const float* __restrict__ wveg, const float* __restrict__ wag,
    const float* __restrict__ QKV,
    u16* __restrict__ QB, u16* __restrict__ KB, u16* __restrict__ VB,
    float* __restrict__ AG) {
  const int bt = blockIdx.x;
  const int b = bt / T_DIM, t = bt % T_DIM;
  const int wave = threadIdx.x >> 5, lane = threadIdx.x & 31;
  const int hk = wave;  // kv head

  const float* xrow = x + (size_t)bt * C_DIM;
  const float* qkvrow = QKV + (size_t)bt * QKV_N;

  // ve gate (VE_GATE_CH=32: one channel per lane)
  float gve = wave_sum(xrow[lane] * wveg[lane * NKV + hk]);
  gve = 2.0f / (1.0f + __expf(-gve));

  // attn gates (ATTN_GATE_CH=12)
#pragma unroll
  for (int s = 0; s < 2; ++s) {
    int h = 2 * wave + s;
    float pa = (lane < 12) ? xrow[lane] * wag[lane * NH + h] : 0.0f;
    pa = wave_sum(pa);
    if (lane == 0) AG[(size_t)bt * NH + h] = 1.0f / (1.0f + __expf(-pa));
  }

  const float* cr = cosp + (size_t)t * (HD / 2);
  const float* sr = sinp + (size_t)t * (HD / 2);
  float c0 = cr[lane], c1 = cr[lane + 32];
  float s0 = sr[lane], s1 = sr[lane + 32];

  // q heads 2w, 2w+1: RoPE then RMS-norm
#pragma unroll
  for (int s = 0; s < 2; ++s) {
    int h = 2 * wave + s;
    const float* qp = qkvrow + h * HD;
    float a0 = qp[lane], a1 = qp[lane + 32];
    float b0 = qp[64 + lane], b1 = qp[96 + lane];
    float r0 = a0 * c0 + b0 * s0, r1 = a1 * c1 + b1 * s1;
    float u0 = -a0 * s0 + b0 * c0, u1 = -a1 * s1 + b1 * c1;
    float ss = wave_sum(r0 * r0 + r1 * r1 + u0 * u0 + u1 * u1);
    float rs = rsqrtf(ss * (1.0f / HD) + 1.1920929e-7f);
    u16* dst = QB + ((size_t)bt * NH + h) * HD;
    dst[lane] = f2bf(r0 * rs);      dst[lane + 32] = f2bf(r1 * rs);
    dst[64 + lane] = f2bf(u0 * rs); dst[96 + lane] = f2bf(u1 * rs);
  }

  // k head hk: second half comes from token t-1 (t==0 -> itself)
  {
    const float* kp = qkvrow + 2048 + hk * HD;
    int tp = (t == 0) ? 0 : t - 1;
    const float* kpp = QKV + ((size_t)(b * T_DIM + tp)) * QKV_N + 2048 + hk * HD;
    float a0 = kp[lane], a1 = kp[lane + 32];
    float b0 = kpp[64 + lane], b1 = kpp[96 + lane];
    float r0 = a0 * c0 + b0 * s0, r1 = a1 * c1 + b1 * s1;
    float u0 = -a0 * s0 + b0 * c0, u1 = -a1 * s1 + b1 * c1;
    float ss = wave_sum(r0 * r0 + r1 * r1 + u0 * u0 + u1 * u1);
    float rs = rsqrtf(ss * (1.0f / HD) + 1.1920929e-7f);
    u16* dst = KB + ((size_t)bt * NKV + hk) * HD;
    dst[lane] = f2bf(r0 * rs);      dst[lane + 32] = f2bf(r1 * rs);
    dst[64 + lane] = f2bf(u0 * rs); dst[96 + lane] = f2bf(u1 * rs);
  }

  // v head hk: v + gate * ve
  {
    const float* vp = qkvrow + 3072 + hk * HD;
    const float* vep = ve + (size_t)bt * (NKV * HD) + hk * HD;
    u16* dst = VB + ((size_t)bt * NKV + hk) * HD;
#pragma unroll
    for (int i = 0; i < 4; ++i) {
      int d = lane + 32 * i;
      dst[d] = f2bf(vp[d] + gve * vep[d]);
    }
  }
}

// ---------------- flash attention, sliding window, WMMA QK^T and PV ----------
#define LQ 136   // 128 + 8 pad
#define LV 72    // 64 + 8 pad
__global__ __launch_bounds__(128) void attn_fwd(
    const u16* __restrict__ QB, const u16* __restrict__ KB,
    const u16* __restrict__ VB, const float* __restrict__ AG,
    u16* __restrict__ YB, const int* __restrict__ wptr) {
  const int q0 = blockIdx.x * 64;
  const int h = blockIdx.y;
  const int b = blockIdx.z;
  const int hk = h >> 1;
  const int win = wptr[0];
  const int tid = threadIdx.x, wave = tid >> 5, lane = tid & 31;
  const int lh = lane >> 4, nl = lane & 15;

  __shared__ __align__(16) u16 sQ[64 * LQ];
  __shared__ __align__(16) u16 sK[64 * LQ];
  __shared__ __align__(16) u16 sVt[128 * LV];       // transposed: [d][key]
  __shared__ __align__(16) u16 sP[4][16 * LV];      // per-wave P tile

  {  // load Q tile 64x128 via async DMA (waited before first use below)
    int row = tid >> 1, seg = (tid & 1) * 64;
    const u16* src = QB + ((size_t)(b * T_DIM + q0 + row) * NH + h) * HD + seg;
    u32 dst = lds_off(&sQ[row * LQ + seg]);
#pragma unroll
    for (int i = 0; i < 8; ++i)
      async_copy_b128(dst + i * 16, src + i * 8);
  }

  float m_i[8], l_i[8];
  v8f O[8];
#pragma unroll
  for (int j = 0; j < 8; ++j) { m_i[j] = -1e30f; l_i[j] = 0.0f; O[j] = zero8(); }

  int kb0 = q0 - win; if (kb0 < 0) kb0 = 0; kb0 = (kb0 / 64) * 64;

  for (int kb = kb0; kb <= q0; kb += 64) {
    {  // K tile 64x128 via async DMA
      int row = tid >> 1, seg = (tid & 1) * 64;
      const u16* src = KB + ((size_t)(b * T_DIM + kb + row) * NKV + hk) * HD + seg;
      u32 dst = lds_off(&sK[row * LQ + seg]);
#pragma unroll
      for (int i = 0; i < 8; ++i)
        async_copy_b128(dst + i * 16, src + i * 8);
    }
    {  // V tile 64x128, stored transposed sVt[d][key] (transpose -> VGPR path)
      int key = tid >> 1, seg = (tid & 1) * 64;
      const u16* src = VB + ((size_t)(b * T_DIM + kb + key) * NKV + hk) * HD + seg;
#pragma unroll
      for (int i = 0; i < 64; ++i) sVt[(seg + i) * LV + key] = src[i];
    }
    async_wait0();      // covers Q tile (first iter) + this K tile
    __syncthreads();

    // S = Q K^T (16 q-rows per wave x 64 keys), contraction over 128 head dims
    v8f S[4];
#pragma unroll
    for (int j = 0; j < 4; ++j) S[j] = zero8();
#pragma unroll
    for (int ks = 0; ks < 4; ++ks) {
      v16bf aQ = frag_a(&sQ[(wave * 16 + nl) * LQ + ks * 32], lh);
#pragma unroll
      for (int j = 0; j < 4; ++j) {
        v16bf bK = frag_b(&sK[(j * 16 + nl) * LQ + ks * 32], lh);
        S[j] = wmma_bf16(aQ, bK, S[j]);
      }
    }

    // scale + sliding-window causal mask; per-row block max
    float mb[8];
#pragma unroll
    for (int r = 0; r < 8; ++r) mb[r] = -1e30f;
#pragma unroll
    for (int j = 0; j < 4; ++j)
#pragma unroll
      for (int r = 0; r < 8; ++r) {
        int qr = q0 + wave * 16 + r + 8 * lh;
        int kr = kb + j * 16 + nl;
        int dist = qr - kr;
        float s = S[j][r] * SCALE_ATTN;
        s = (dist >= 0 && dist <= win) ? s : -1e30f;
        S[j][r] = s;
        mb[r] = fmaxf(mb[r], s);
      }
#pragma unroll
    for (int r = 0; r < 8; ++r)
#pragma unroll
      for (int m = 8; m; m >>= 1) mb[r] = fmaxf(mb[r], __shfl_xor(mb[r], m, 32));

    float al[8], rs[8];
#pragma unroll
    for (int r = 0; r < 8; ++r) {
      float mn = fmaxf(m_i[r], mb[r]);
      al[r] = __expf(m_i[r] - mn);
      m_i[r] = mn;
      rs[r] = 0.0f;
    }
#pragma unroll
    for (int j = 0; j < 4; ++j)
#pragma unroll
      for (int r = 0; r < 8; ++r) {
        float p = __expf(S[j][r] - m_i[r]);
        S[j][r] = p;
        rs[r] += p;
      }
#pragma unroll
    for (int r = 0; r < 8; ++r) {
#pragma unroll
      for (int m = 8; m; m >>= 1) rs[r] += __shfl_xor(rs[r], m, 32);
      l_i[r] = l_i[r] * al[r] + rs[r];
    }
#pragma unroll
    for (int c = 0; c < 8; ++c)
#pragma unroll
      for (int r = 0; r < 8; ++r) O[c][r] *= al[r];

    // C/D layout -> A layout via per-wave LDS round trip
#pragma unroll
    for (int j = 0; j < 4; ++j)
#pragma unroll
      for (int r = 0; r < 8; ++r)
        sP[wave][(r + 8 * lh) * LV + j * 16 + nl] = f2bf(S[j][r]);

    // O += P V  (contraction over 64 keys)
#pragma unroll
    for (int ks = 0; ks < 2; ++ks) {
      v16bf aP = frag_a(&sP[wave][nl * LV + ks * 32], lh);
#pragma unroll
      for (int c = 0; c < 8; ++c) {
        v16bf bV = frag_b(&sVt[(c * 16 + nl) * LV + ks * 32], lh);
        O[c] = wmma_bf16(aP, bV, O[c]);
      }
    }
    __syncthreads();
  }

  // epilogue: attn gate, 1/l, write bf16 Y (layout [B*T, NH*HD])
#pragma unroll
  for (int r = 0; r < 8; ++r) {
    int qr = q0 + wave * 16 + r + 8 * lh;
    float g = AG[(size_t)(b * T_DIM + qr) * NH + h];
    float inv = g / l_i[r];
    u16* dst = YB + (size_t)(b * T_DIM + qr) * C_DIM + h * HD;
#pragma unroll
    for (int c = 0; c < 8; ++c) dst[c * 16 + nl] = f2bf(O[c][r] * inv);
  }
}

// ---------------- launch ------------------------------------------------------
extern "C" void kernel_launch(void* const* d_in, const int* in_sizes, int n_in,
                              void* d_out, int out_size, void* d_ws, size_t ws_size,
                              hipStream_t stream) {
  (void)in_sizes; (void)n_in; (void)out_size; (void)ws_size;
  const float* x    = (const float*)d_in[0];
  const float* ve   = (const float*)d_in[1];
  const float* cosp = (const float*)d_in[2];
  const float* sinp = (const float*)d_in[3];
  const float* wq   = (const float*)d_in[4];
  const float* wk   = (const float*)d_in[5];
  const float* wv   = (const float*)d_in[6];
  const float* wpr  = (const float*)d_in[7];
  const float* wveg = (const float*)d_in[8];
  const float* wag  = (const float*)d_in[9];
  const float* pscal= (const float*)d_in[10];
  const int*   wsz  = (const int*)d_in[11];

  char* ws = (char*)d_ws;
  const size_t SZ_XB   = (size_t)M_ROWS * C_DIM * 2;          // 16 MiB
  const size_t SZ_WQKV = (size_t)C_DIM * QKV_N * 2;           // 16 MiB
  const size_t SZ_WPR  = (size_t)C_DIM * C_DIM * 2;           //  8 MiB
  const size_t SZ_QKV  = (size_t)M_ROWS * QKV_N * 4;          // 64 MiB
  const size_t SZ_QB   = (size_t)M_ROWS * NH * HD * 2;        // 16 MiB
  const size_t SZ_KB   = (size_t)M_ROWS * NKV * HD * 2;       //  8 MiB
  const size_t SZ_VB   = SZ_KB;

  u16*   XB    = (u16*)(ws);
  u16*   WQKV  = (u16*)(ws + SZ_XB);
  u16*   WPROJ = (u16*)(ws + SZ_XB + SZ_WQKV);
  float* QKVf  = (float*)(ws + SZ_XB + SZ_WQKV + SZ_WPR);
  u16*   QB    = (u16*)(ws + SZ_XB + SZ_WQKV + SZ_WPR + SZ_QKV);
  u16*   KB    = (u16*)((char*)QB + SZ_QB);
  u16*   VB    = (u16*)((char*)KB + SZ_KB);
  float* AG    = (float*)((char*)VB + SZ_VB);
  u16*   YB    = XB;  // reuse: x(bf16) dead after QKV GEMM

  const int TPB = 256;
  auto blocks = [&](long n) { return (unsigned)((n + TPB - 1) / TPB); };

  // convert inputs/weights to bf16
  cvt_f32_bf16<<<blocks((long)M_ROWS * C_DIM), TPB, 0, stream>>>(x,   XB,    M_ROWS, C_DIM, C_DIM, 0);
  cvt_f32_bf16<<<blocks((long)C_DIM * 2048),   TPB, 0, stream>>>(wq,  WQKV,  C_DIM, 2048, QKV_N, 0);
  cvt_f32_bf16<<<blocks((long)C_DIM * 1024),   TPB, 0, stream>>>(wk,  WQKV,  C_DIM, 1024, QKV_N, 2048);
  cvt_f32_bf16<<<blocks((long)C_DIM * 1024),   TPB, 0, stream>>>(wv,  WQKV,  C_DIM, 1024, QKV_N, 3072);
  cvt_f32_bf16<<<blocks((long)C_DIM * C_DIM),  TPB, 0, stream>>>(wpr, WPROJ, C_DIM, C_DIM, C_DIM, 0);

  // QKV = x @ [wq|wk|wv]
  gemm_bf16<<<dim3(QKV_N / 128, M_ROWS / 128), 128, 0, stream>>>(
      XB, WQKV, QKVf, M_ROWS, QKV_N, C_DIM, nullptr);

  // gates + RoPE + k-shift + rmsnorm -> bf16 Q/K/V
  qkv_prep<<<M_ROWS, 256, 0, stream>>>(x, ve, cosp, sinp, wveg, wag, QKVf,
                                       QB, KB, VB, AG);

  // sliding-window flash attention
  attn_fwd<<<dim3(T_DIM / 64, NH, B_DIM), 128, 0, stream>>>(QB, KB, VB, AG, YB, wsz);

  // out = (Y @ wproj) * (1 + proj_scalar)
  gemm_bf16<<<dim3(C_DIM / 128, M_ROWS / 128), 128, 0, stream>>>(
      YB, WPROJ, (float*)d_out, M_ROWS, C_DIM, C_DIM, pscal);
}
